// GraphFilter_76562087019000
// MI455X (gfx1250) — matrix operand order repaired and checked by reading, not compiled, not run on hardware
//
#include <hip/hip_runtime.h>
#include <stdint.h>

// GraphFilter: y = ReLU( sum_{k=0}^{7} h_k * S^k X )
// S: 8192x8192 fp32, X: 8192x256 fp32. Convert S to bf16 once (128MB -> fits
// 192MB L2), chain 7 bf16 WMMA GEMMs with fp32 accumulation. K-tiles staged
// into LDS via async global->LDS DMA (ASYNCcnt), 4-deep pipelined.

#define NN 8192
#define FF 256
#define ORDER 8
#define KT 32          // K tile depth
#define PITCH 40       // padded bf16 elems per LDS row/col (80 B, 16B aligned)
#define NITER (NN / KT)
#define NBUF 4         // LDS pipeline depth (issue distance 3)

typedef __attribute__((ext_vector_type(16))) __bf16 v16bf;
typedef __attribute__((ext_vector_type(8)))  float  v8f;
typedef __attribute__((ext_vector_type(4)))  int    v4i;

struct Frag32B { uint4 lo, hi; };
static_assert(sizeof(Frag32B) == 32, "frag size");
static_assert(sizeof(v16bf) == 32, "v16bf size");

struct US8 { uint16_t v[8]; };
static_assert(sizeof(US8) == 16, "us8 size");

typedef __attribute__((address_space(1))) v4i* gv4i_p;   // global v4i*
typedef __attribute__((address_space(3))) v4i* lv4i_p;   // LDS v4i*

// 16B global -> LDS copy: async DMA when available, sync fallback otherwise.
static __device__ __forceinline__ void cp16_g2l(void* lds, const void* g) {
#if __has_builtin(__builtin_amdgcn_global_load_async_to_lds_b128)
  __builtin_amdgcn_global_load_async_to_lds_b128((gv4i_p)g, (lv4i_p)lds, 0, 0);
#else
  *(uint4*)lds = *(const uint4*)g;
#endif
}

#if __has_builtin(__builtin_amdgcn_s_wait_asynccnt)
#define WAITA(n) __builtin_amdgcn_s_wait_asynccnt(n)
#else
#define WAITA(n) asm volatile("s_wait_asynccnt " #n ::: "memory")
#endif

static __device__ __forceinline__ v16bf load_frag(const void* p0, const void* p1) {
  Frag32B f;
  f.lo = *(const uint4*)p0;   // 8 bf16
  f.hi = *(const uint4*)p1;   // 8 bf16
  return __builtin_bit_cast(v16bf, f);
}

// fp32 -> bf16, round-to-nearest-even
static __device__ __forceinline__ uint16_t f2bf(float x) {
  uint32_t u = __builtin_bit_cast(uint32_t, x);
  uint32_t r = u + 0x7FFFu + ((u >> 16) & 1u);
  return (uint16_t)(r >> 16);
}

// ---------------- S fp32 -> bf16 (row-major kept) ----------------
__global__ __launch_bounds__(256) void convS(const float4* __restrict__ in,
                                             ushort4* __restrict__ out) {
  size_t i = (size_t)blockIdx.x * blockDim.x + threadIdx.x;
  float4 v = in[i];
  ushort4 o;
  o.x = f2bf(v.x); o.y = f2bf(v.y); o.z = f2bf(v.z); o.w = f2bf(v.w);
  out[i] = o;
}

// ------- init: d_out = h0 * X ; Z0(col-major FxN, bf16) = X^T -------
__global__ __launch_bounds__(256) void initX(const float* __restrict__ X,
                                             const float* __restrict__ coeff,
                                             float* __restrict__ out,
                                             uint16_t* __restrict__ Zcm) {
  int idx = blockIdx.x * 256 + threadIdx.x;   // idx over N*F
  int m = idx >> 8;        // row (node)
  int f = idx & (FF - 1);  // channel
  float x = X[idx];
  out[idx] = coeff[0] * x;
  Zcm[(size_t)f * NN + m] = f2bf(x);
}

// ---------------- one filter tap: C = S_bf16 @ Zin ----------------
// grid (NN/64, FF/128), 256 threads = 8 wave32s.
// WG tile: 64 rows x 128 cols. wave (strip=w&3, colhalf=w>>2) owns
// rows [strip*16,+16) x cols [colhalf*64,+64) = 4 WMMA tiles.
// Each K-tile (64x32 A, 128x32 B^T) staged once per WG into LDS by async DMA,
// 4 buffers deep so the DMA for tile t+3 overlaps compute on tile t.
__global__ __launch_bounds__(256) void gemm_step(const uint16_t* __restrict__ S,
                                                 const uint16_t* __restrict__ Zin,
                                                 uint16_t* __restrict__ Znext,
                                                 float* __restrict__ acc_out,
                                                 const float* __restrict__ coeff,
                                                 int k, int last) {
  constexpr int ABUF = 64 * PITCH * 2;              // 5120 B per A buffer
  constexpr int BBUF = 128 * PITCH * 2;             // 10240 B per B buffer
  constexpr int BUFSZ = ABUF + BBUF;                // 15360 B per stage
  __shared__ alignas(16) uint8_t smem[NBUF * BUFSZ];  // 60 KB
  uint8_t* As = smem;              // A at offset 0 of each stage buffer
  uint8_t* Bs = smem + ABUF;       // B at offset ABUF of each stage buffer

  const int tid  = threadIdx.x;
  const int wave = tid >> 5;
  const int lane = tid & 31;
  const int strip   = wave & 3;
  const int colhalf = wave >> 2;

  const int m0 = blockIdx.x * 64;
  const int f0 = blockIdx.y * 128;

  const int laneRow = lane & 15;   // A row / B col / C col within 16x16 tile
  const int hi      = lane >> 4;
  const int hiA8    = hi * 8;      // A: lanes 16-31 hold K group +8
  const int hiB8    = hi * 8;      // B: lanes 16-31 hold K group +8
  const int hiC8    = hi * 8;      // C/D: lanes 16-31 hold M rows 8..15

  const float h = coeff[k];

  // ---- staging assignments (per thread: 1 A chunk + 2 B chunks of 16B) ----
  const int aRow = tid >> 2, aSeg = tid & 3;                 // 64 rows x 4 segs
  const uint16_t* gA = S + (size_t)(m0 + aRow) * NN + aSeg * 8;
  uint8_t* lA = As + aRow * (PITCH * 2) + aSeg * 16;

  const int bCol0 = tid >> 1, bSeg0 = (tid & 1) * 2;         // 16B chunk 0 or 2 of cols 0..127
  const uint16_t* gB0 = Zin + (size_t)(f0 + bCol0) * NN + bSeg0 * 8;
  uint8_t* lB0 = Bs + bCol0 * (PITCH * 2) + bSeg0 * 16;
  const int bCol1 = bCol0 + 64;                              // cols 64..191, same chunk
  const uint16_t* gB1 = Zin + (size_t)(f0 + bCol1) * NN + bSeg0 * 8;
  uint8_t* lB1 = Bs + bCol1 * (PITCH * 2) + bSeg0 * 16;

  // ---- per-lane fragment base addresses in LDS ----
  const uint8_t* aFrag = As + (strip * 16 + laneRow) * (PITCH * 2) + hiA8 * 2;
  const uint8_t* bFrag[4];
#pragma unroll
  for (int t = 0; t < 4; ++t)
    bFrag[t] = Bs + (colhalf * 64 + t * 16 + laneRow) * (PITCH * 2) + hiB8 * 2;

  v8f acc[4] = {};

  // stage tile `t` into LDS stage buffer `t % NBUF` (3 async b128 per thread)
  auto stage = [&](int t) {
    const int off = (t & (NBUF - 1)) * BUFSZ;
    const int kk  = t * KT;
    cp16_g2l(lA + off, gA + kk);
    cp16_g2l(lB0 + off, gB0 + kk);
    cp16_g2l(lB1 + off, gB1 + kk);
  };

  // 4 WMMAs on LDS stage buffer `t % NBUF`
  auto compute = [&](int t) {
    const int off = (t & (NBUF - 1)) * BUFSZ;
    const uint8_t* ap = aFrag + off;
    v16bf a = load_frag(ap, ap + 32);
#pragma unroll
    for (int tt = 0; tt < 4; ++tt) {
      const uint8_t* bp = bFrag[tt] + off;
      v16bf b = load_frag(bp, bp + 32);
      acc[tt] = __builtin_amdgcn_wmma_f32_16x16x32_bf16(
          /*neg_a=*/false, a, /*neg_b=*/false, b,
          /*c_mod=*/(short)0, acc[tt], /*reuse_a=*/false, /*reuse_b=*/false);
    }
  };

  // prologue: fill pipeline with tiles 0..2
  stage(0);
  stage(1);
  stage(2);

  // steady state: unconditional stage of t+3, wait for group t (<= 3 groups
  // of 3 DMAs in flight beyond it), barrier, compute, barrier (buffer reuse).
  // unroll(disable): a single body keeps the accumulators in-place (D==C),
  // avoiding the cross-body v_mov/v_nop shuffle the unroller introduces.
#pragma clang loop unroll(disable)
  for (int t = 0; t < NITER - 3; ++t) {
    stage(t + 3);
    WAITA(9);
    __syncthreads();
    compute(t);
    __syncthreads();
  }
  // drain: tiles NITER-3 .. NITER-1
  WAITA(6); __syncthreads(); compute(NITER - 3); __syncthreads();
  WAITA(3); __syncthreads(); compute(NITER - 2); __syncthreads();
  WAITA(0); __syncthreads(); compute(NITER - 1);

  // Epilogue: acc_out += h * C (ReLU on last tap); Znext = bf16(C) col-major.
  const int m_wave = m0 + strip * 16;
#pragma unroll
  for (int t = 0; t < 4; ++t) {
    const int f = f0 + colhalf * 64 + t * 16 + laneRow;
    float* op = acc_out + (size_t)(m_wave + hiC8) * FF + f;
    US8 zb;
#pragma unroll
    for (int r = 0; r < 8; ++r) {
      float c = acc[t][r];
      float v = op[(size_t)r * FF] + h * c;
      if (last) v = fmaxf(v, 0.0f);
      op[(size_t)r * FF] = v;
      zb.v[r] = f2bf(c);
    }
    if (!last) {
      // lane writes 8 consecutive rows of its column: 16B contiguous store
      *(uint4*)(Znext + (size_t)f * NN + m_wave + hiC8) = __builtin_bit_cast(uint4, zb);
    }
  }
}

extern "C" void kernel_launch(void* const* d_in, const int* in_sizes, int n_in,
                              void* d_out, int out_size, void* d_ws, size_t ws_size,
                              hipStream_t stream) {
  (void)in_sizes; (void)n_in; (void)out_size; (void)ws_size;
  const float* X     = (const float*)d_in[0];   // [N, F] fp32
  const float* gso   = (const float*)d_in[1];   // [N, N] fp32
  const float* coeff = (const float*)d_in[2];   // [ORDER] fp32
  float* out = (float*)d_out;                   // [N, F] fp32

  uint8_t* ws = (uint8_t*)d_ws;
  uint16_t* Sb = (uint16_t*)ws;                                // 128 MB bf16 S
  uint16_t* Z0 = (uint16_t*)(ws + (size_t)NN * NN * 2);        // 4 MB bf16 Z ping
  uint16_t* Z1 = Z0 + (size_t)NN * FF;                         // 4 MB bf16 Z pong

  // 1) S -> bf16 (one-time per call; thereafter S lives in the 192MB L2)
  convS<<<(unsigned)(((size_t)NN * NN / 4) / 256), 256, 0, stream>>>(
      (const float4*)gso, (ushort4*)Sb);

  // 2) d_out = h0 * X ; Z0 = bf16(X) column-major
  initX<<<(NN * FF) / 256, 256, 0, stream>>>(X, coeff, out, Z0);

  // 3) taps k = 1..7: Z_k = S @ Z_{k-1}; d_out += h_k * Z_k; ReLU on last
  uint16_t* zi = Z0;
  uint16_t* zo = Z1;
  for (int k = 1; k < ORDER; ++k) {
    gemm_step<<<dim3(NN / 64, FF / 128), 256, 0, stream>>>(
        Sb, zi, zo, out, coeff, k, (k == ORDER - 1) ? 1 : 0);
    uint16_t* t = zi; zi = zo; zo = t;
  }
}
